// VectorQuantizer_76158360093050
// MI455X (gfx1250) — matrix operand order repaired and checked by reading, not compiled, not run on hardware
//
#include <hip/hip_runtime.h>
#include <hip/hip_bf16.h>

// ---------------------------------------------------------------------------
// VQ (masked-cdist argmin + EMA codebook update) for MI455X / gfx1250.
//
// Strategy: bin atoms by element (hist/scan/scatter), then per 4-element
// group (16 codes) run 16-atom tiles through v_wmma_f32_16x16x32_f16 to get
// all atom x code dot products; argmin over each atom's own 4 columns of
// (|c|^2 - 2 h.c). Memory-bound: ~272MB traffic => ~12us at 23.3 TB/s.
// ---------------------------------------------------------------------------

typedef __attribute__((ext_vector_type(16))) _Float16 v16h;
typedef __attribute__((ext_vector_type(8)))  float    v8f;

#define NE 100
#define NC 400
#define ND 128
#define NN 262144
#define NG 25            // groups of 4 elements (16 codes each)
#define GAMMA 0.99f
#define BETA  0.25f
#define SPLIT 32         // blocks per group
#define WAVES 4
#define TPB   (WAVES * 32)

__global__ void k_zero(float* wsf, int nwords) {
  int i = blockIdx.x * blockDim.x + threadIdx.x;
  if (i < nwords) wsf[i] = 0.0f;
}

__global__ void k_hist(const int* __restrict__ ei, int* __restrict__ ec) {
  int i = blockIdx.x * blockDim.x + threadIdx.x;
  if (i < NN) atomicAdd(&ec[ei[i]], 1);
}

__global__ void k_scan(const int* __restrict__ ec, int* __restrict__ off,
                       int* __restrict__ cur) {
  if (threadIdx.x == 0) {
    int run = 0;
    for (int e = 0; e < NE; ++e) { off[e] = run; cur[e] = run; run += ec[e]; }
    off[NE] = run;
  }
}

__global__ void k_scatter(const int* __restrict__ ei, int* __restrict__ cur,
                          int* __restrict__ sorted) {
  int i = blockIdx.x * blockDim.x + threadIdx.x;
  if (i < NN) {
    int slot = atomicAdd(&cur[ei[i]], 1);
    sorted[slot] = i;
  }
}

__global__ __launch_bounds__(TPB)
void k_vq(const float* __restrict__ h, const int* __restrict__ ei,
          const float* __restrict__ cb, const int* __restrict__ off,
          const int* __restrict__ sorted,
          float* __restrict__ hq, float* __restrict__ at_out,
          float* __restrict__ code_counts, float* __restrict__ code_sums,
          float* __restrict__ loss_out) {
  __shared__ _Float16 cbh[16][136];          // group codebook, f16, padded
  __shared__ float    cnorm[16];
  __shared__ float    sums16[16][ND];        // per-block segment sums
  __shared__ float    counts16[16];
  __shared__ float    lossblk;
  __shared__ float    htile[WAVES][16][136]; // 136*4B = 544B row (16B aligned)
  __shared__ float    score[WAVES][16][17];
  __shared__ int      aid[WAVES][16];
  __shared__ int      rowcb[WAVES][16];
  __shared__ int      colbest[WAVES][16];

  const int g    = blockIdx.x / SPLIT;
  const int s    = blockIdx.x % SPLIT;
  const int tid  = threadIdx.x;
  const int w    = tid >> 5;
  const int lane = tid & 31;
  const int q    = lane & 15;   // A-row / B,C-column owned by this lane
  const int hf   = lane >> 4;   // lane-half (K split)

  // ---- block init: group codebook (f16) + norms + accumulators ----
  for (int i = tid; i < 16 * ND; i += TPB) {
    int c = i >> 7, d = i & 127;
    cbh[c][d] = (_Float16)cb[(size_t)(g * 16 + c) * ND + d];
    ((float*)sums16)[i] = 0.0f;
  }
  if (tid < 16) {
    float sacc = 0.0f;
    const float* p = &cb[(size_t)(g * 16 + tid) * ND];
    for (int d = 0; d < ND; ++d) sacc += p[d] * p[d];
    cnorm[tid] = sacc;
    counts16[tid] = 0.0f;
  }
  if (tid == 0) lossblk = 0.0f;
  __syncthreads();

  // ---- B fragments: column q = code g*16+q, K contiguous (k = 16*hf + elem), same as A ----
  v16h Bfrag[4];
#pragma unroll
  for (int c = 0; c < 4; ++c)
#pragma unroll
    for (int v = 0; v < 8; ++v) {
      int k = c * 32 + hf * 16 + 2 * v;
      Bfrag[c][2 * v]     = cbh[q][k];
      Bfrag[c][2 * v + 1] = cbh[q][k + 1];
    }

  float lossLocal = 0.0f;
  const int lo = off[4 * g];
  const int hiR = off[4 * g + 4];
  const int len = hiR - lo;

  for (int t = s * WAVES + w; t * 16 < len; t += SPLIT * WAVES) {
    const int base = lo + t * 16;
    const int nrow = min(16, hiR - base);

    if (lane < 16) {
      int slot = base + min(lane, nrow - 1);
      int a = sorted[slot];
      aid[w][lane] = a;
      rowcb[w][lane] = (ei[a] - 4 * g) * 4;  // candidate column base
    }
    __builtin_amdgcn_wave_barrier();

    // stage 16 h rows, coalesced (lane-parallel over D)
    for (int r = 0; r < 16; ++r) {
      const float4 hv = *(const float4*)&h[(size_t)aid[w][r] * ND + lane * 4];
      *(float4*)&htile[w][r][lane * 4] = hv;
    }
    __builtin_amdgcn_wave_barrier();

    // ---- dots = h_tile(16x128) x codes(128x16) via 4 chained WMMAs ----
    v8f acc = {0.f, 0.f, 0.f, 0.f, 0.f, 0.f, 0.f, 0.f};
#pragma unroll
    for (int c = 0; c < 4; ++c) {
      v16h A;
#pragma unroll
      for (int v = 0; v < 8; ++v) {
        int d = c * 32 + hf * 16 + 2 * v;
        A[2 * v]     = (_Float16)htile[w][q][d];
        A[2 * v + 1] = (_Float16)htile[w][q][d + 1];
      }
      acc = __builtin_amdgcn_wmma_f32_16x16x32_f16(
          false, A, false, Bfrag[c], (short)0, acc, false, false);
    }

    // scores: |c_n|^2 - 2*dot (|h|^2 constant per row, drops from argmin)
    const float cn = cnorm[q];
#pragma unroll
    for (int r = 0; r < 8; ++r) score[w][r + hf * 8][q] = cn - 2.0f * acc[r];
    __builtin_amdgcn_wave_barrier();

    if (lane < nrow) {
      const int cbase = rowcb[w][lane];
      float best = score[w][lane][cbase];
      int bj = 0;
#pragma unroll
      for (int j = 1; j < 4; ++j) {
        float v2 = score[w][lane][cbase + j];
        if (v2 < best) { best = v2; bj = j; }
      }
      const int col = cbase + bj;
      colbest[w][lane] = col;
      at_out[aid[w][lane]] = (float)(g * 16 + col);
      atomicAdd(&counts16[col], 1.0f);
    }
    __builtin_amdgcn_wave_barrier();

    // z_q writeback (fp32 from global codebook), loss, segment sums
    for (int r = 0; r < nrow; ++r) {
      const int col = colbest[w][r];
      const int a = aid[w][r];
      const float4 zq = *(const float4*)&cb[(size_t)(g * 16 + col) * ND + lane * 4];
      const float4 hv = *(const float4*)&htile[w][r][lane * 4];
      *(float4*)&hq[(size_t)a * ND + lane * 4] = zq;
      float dx = hv.x - zq.x, dy = hv.y - zq.y;
      float dz = hv.z - zq.z, dw = hv.w - zq.w;
      lossLocal += dx * dx + dy * dy + dz * dz + dw * dw;
      atomicAdd(&sums16[col][lane * 4 + 0], hv.x);
      atomicAdd(&sums16[col][lane * 4 + 1], hv.y);
      atomicAdd(&sums16[col][lane * 4 + 2], hv.z);
      atomicAdd(&sums16[col][lane * 4 + 3], hv.w);
    }
  }

  atomicAdd(&lossblk, lossLocal);
  __syncthreads();

  // flush block accumulators (code_sums is contiguous per group)
  for (int i = tid; i < 16 * ND; i += TPB) {
    float v2 = ((float*)sums16)[i];
    if (v2 != 0.0f) atomicAdd(&code_sums[(size_t)g * 16 * ND + i], v2);
  }
  if (tid < 16) {
    float cv = counts16[tid];
    if (cv != 0.0f) atomicAdd(&code_counts[g * 16 + tid], cv);
  }
  if (tid == 0) atomicAdd(loss_out, lossblk);
}

__global__ void k_ema(const float* __restrict__ cbook,
                      const float* __restrict__ ema_count,
                      const float* __restrict__ ema_sum,
                      const int* __restrict__ ec,
                      const float* __restrict__ code_counts,
                      const float* __restrict__ code_sums,
                      const float* __restrict__ loss_sum,
                      float* __restrict__ ncb, float* __restrict__ ncnt,
                      float* __restrict__ nsum, float* __restrict__ vloss) {
  int idx = blockIdx.x * blockDim.x + threadIdx.x;
  if (idx == 0) *vloss = BETA * (*loss_sum) / (float)((size_t)NN * ND);
  if (idx >= NC * ND) return;
  int c = idx >> 7;
  int e = c >> 2;
  bool present = ec[e] > 0;
  float ncount = present ? (GAMMA * ema_count[c] + (1.0f - GAMMA) * code_counts[c])
                         : ema_count[c];
  float ns = present ? (GAMMA * ema_sum[idx] + (1.0f - GAMMA) * code_sums[idx])
                     : ema_sum[idx];
  nsum[idx] = ns;
  ncb[idx] = present ? (ns / fmaxf(ncount, 1e-5f)) : cbook[idx];
  if ((idx & 127) == 0) ncnt[c] = ncount;
}

extern "C" void kernel_launch(void* const* d_in, const int* in_sizes, int n_in,
                              void* d_out, int out_size, void* d_ws, size_t ws_size,
                              hipStream_t stream) {
  const float* h         = (const float*)d_in[0];
  const int*   ei        = (const int*)d_in[1];
  const float* cb        = (const float*)d_in[2];
  const float* ema_count = (const float*)d_in[3];
  const float* ema_sum   = (const float*)d_in[4];

  // outputs: h_q[N,D], atom_types[N], vq_loss[1], new_codebook[C,D],
  //          new_count[C], new_sum[C,D]  (flat concat, float32)
  float* out   = (float*)d_out;
  float* hq    = out;
  float* at    = hq + (size_t)NN * ND;
  float* vloss = at + NN;
  float* ncb   = vloss + 1;
  float* ncnt  = ncb + (size_t)NC * ND;
  float* nsum  = ncnt + NC;

  // workspace layout (32-bit words)
  float* wsf = (float*)d_ws;
  int*   wsi = (int*)d_ws;
  const int W_EC = 0;                  // elem_counts [104]
  const int W_CC = 104;                // code_counts [400]
  const int W_LOSS = 504;              // loss accumulator (pad to 512)
  const int W_CS = 512;                // code_sums [C*D]
  const int W_OFF = W_CS + NC * ND;    // elem_offsets [104]
  const int W_CUR = W_OFF + 104;       // scatter cursors [104]
  const int W_SORT = W_CUR + 104;      // sorted atom ids [N]
  int*   ec     = wsi + W_EC;
  float* ccnt   = wsf + W_CC;
  float* lsum   = wsf + W_LOSS;
  float* csum   = wsf + W_CS;
  int*   off    = wsi + W_OFF;
  int*   cur    = wsi + W_CUR;
  int*   sorted = wsi + W_SORT;

  const int zwords = W_CS + NC * ND;   // counts + sums + loss
  hipLaunchKernelGGL(k_zero, dim3((zwords + 255) / 256), dim3(256), 0, stream,
                     wsf, zwords);
  hipLaunchKernelGGL(k_hist, dim3((NN + 255) / 256), dim3(256), 0, stream,
                     ei, ec);
  hipLaunchKernelGGL(k_scan, dim3(1), dim3(32), 0, stream, ec, off, cur);
  hipLaunchKernelGGL(k_scatter, dim3((NN + 255) / 256), dim3(256), 0, stream,
                     ei, cur, sorted);
  hipLaunchKernelGGL(k_vq, dim3(NG * SPLIT), dim3(TPB), 0, stream,
                     h, ei, cb, off, sorted, hq, at, ccnt, csum, lsum);
  hipLaunchKernelGGL(k_ema, dim3((NC * ND + 255) / 256), dim3(256), 0, stream,
                     cb, ema_count, ema_sum, ec, ccnt, csum, lsum,
                     ncb, ncnt, nsum, vloss);
}
